// SGNS_29248727286474
// MI455X (gfx1250) — matrix operand order repaired and checked by reading, not compiled, not run on hardware
//
#include <hip/hip_runtime.h>

typedef __attribute__((ext_vector_type(16))) _Float16 v16h;
typedef __attribute__((ext_vector_type(8)))  float    v8f;
typedef __attribute__((ext_vector_type(4)))  float    v4f;

#define DIM 128
#define WAVES_PER_BLOCK 8
#define BLOCK_THREADS 256

// ---- WMMA step: D = A(16x32 f16) * B(32x16 f16) + C(16x16 f32) ----
__device__ __forceinline__ v8f wmma_step(v16h a, v16h b, v8f c) {
  return __builtin_amdgcn_wmma_f32_16x16x32_f16(
      /*neg_a=*/false, a, /*neg_b=*/false, b,
      /*c_mod=*/(short)0, c, /*reuse_a=*/false, /*reuse_b=*/false);
}

// A fragment, 16-bit A 16x32 layout (contiguous K, matching B):
// lane<16 : row=lane,    K = c*32 + 0..15
// lane>=16: row=lane-16, K = c*32 + 16..31
__device__ __forceinline__ v16h load_a_frag(const float* __restrict__ row,
                                            int chunk, int half) {
  const float* p = row + chunk * 32 + half * 16;
  const v4f* q0 = (const v4f*)p;          // K base+0..7   (32B aligned)
  const v4f* q1 = (const v4f*)(p + 8);    // K base+8..15
  v4f x0 = q0[0], x1 = q0[1], x2 = q1[0], x3 = q1[1];
  v16h r;
#pragma unroll
  for (int e = 0; e < 4; ++e) {
    r[e]      = (_Float16)x0[e];
    r[4 + e]  = (_Float16)x1[e];
    r[8 + e]  = (_Float16)x2[e];
    r[12 + e] = (_Float16)x3[e];
  }
  return r;
}

// B fragment, 16-bit B 32x16 layout:
// lane<16 : col=lane,    K = c*32 + 0..15
// lane>=16: col=lane-16, K = c*32 + 16..31
__device__ __forceinline__ v16h load_b_frag(const float* __restrict__ row,
                                            int chunk, int half) {
  const float* p = row + chunk * 32 + half * 16;
  const v4f* q = (const v4f*)p;           // 64B aligned
  v4f x0 = q[0], x1 = q[1], x2 = q[2], x3 = q[3];
  v16h r;
#pragma unroll
  for (int e = 0; e < 4; ++e) {
    r[e]      = (_Float16)x0[e];
    r[4 + e]  = (_Float16)x1[e];
    r[8 + e]  = (_Float16)x2[e];
    r[12 + e] = (_Float16)x3[e];
  }
  return r;
}

// Compute diag score v[b_i] . u[b_i] for this lane's group via WMMA diag trick.
__device__ __forceinline__ float wmma_diag_score(const float* __restrict__ urow,
                                                 int half, int li,
                                                 v16h a0, v16h a1, v16h a2, v16h a3) {
  v16h b0 = load_b_frag(urow, 0, half);
  v16h b1 = load_b_frag(urow, 1, half);
  v16h b2 = load_b_frag(urow, 2, half);
  v16h b3 = load_b_frag(urow, 3, half);
  v8f acc = {};
  acc = wmma_step(a0, b0, acc);
  acc = wmma_step(a1, b1, acc);
  acc = wmma_step(a2, b2, acc);
  acc = wmma_step(a3, b3, acc);
  // diag (i,i): lanes 0..7 -> acc[lane]; lanes 24..31 -> acc[lane-24]
  float d = acc[0];
#pragma unroll
  for (int j = 1; j < 8; ++j) d = (li == j) ? acc[j] : d;
  return d;
}

// log(sigmoid(x) + eps) with fast hardware reciprocal (v_rcp_f32, ~1 ulp)
__device__ __forceinline__ float log_sigmoid_eps(float x) {
  float s = __builtin_amdgcn_rcpf(1.0f + __expf(-x));
  return __logf(s + 1e-9f);
}

__global__ void __launch_bounds__(BLOCK_THREADS)
sgns_wmma_kernel(const int* __restrict__ center,
                 const int* __restrict__ pos,
                 const int* __restrict__ neg,
                 const float* __restrict__ in_embed,
                 const float* __restrict__ out_embed,
                 float* __restrict__ partial,
                 int K, int ngroups, float invB, float invBK) {
  const int lane = threadIdx.x & 31;
  const int wave = threadIdx.x >> 5;
  const int group = blockIdx.x * WAVES_PER_BLOCK + wave;

  float local = 0.0f;
  if (group < ngroups) {                 // uniform per wave: EXEC stays all-ones
    const int i    = lane & 15;         // row/col index within 16-group
    const int half = lane >> 4;
    const int b    = group * 16 + i;

    const bool has_diag = (lane < 8) || (lane >= 24);
    const int  li = lane & 7;

    // A = 16 center rows (f32 -> f16), kept in registers for all 11 products
    const float* vrow = in_embed + (size_t)center[b] * DIM;
    v16h a0 = load_a_frag(vrow, 0, half);
    v16h a1 = load_a_frag(vrow, 1, half);
    v16h a2 = load_a_frag(vrow, 2, half);
    v16h a3 = load_a_frag(vrow, 3, half);

    float sum_pos = 0.0f;
    float sum_neg = 0.0f;

    // positive sample: log(sigmoid(score) + eps)
    {
      const float* urow = out_embed + (size_t)pos[b] * DIM;
      float d = wmma_diag_score(urow, half, li, a0, a1, a2, a3);
      if (has_diag) sum_pos = log_sigmoid_eps(d);
    }

    // negative samples: log(sigmoid(-score) + eps)
    const int* nb = neg + (size_t)b * K;
    int cur_idx = nb[0];
    for (int k = 0; k < K; ++k) {
      const float* urow = out_embed + (size_t)cur_idx * DIM;
      if (k + 1 < K) {
        cur_idx = nb[k + 1];             // issue next index load early
        const float* nxt = out_embed + (size_t)cur_idx * DIM;
        __builtin_prefetch(nxt + (lane & 3) * 32, 0, 3);  // pull row into WGP$
      }
      float d = wmma_diag_score(urow, half, li, a0, a1, a2, a3);
      if (has_diag) sum_neg += log_sigmoid_eps(-d);
    }

    local = invB * sum_pos + invBK * sum_neg;
  }

  // deterministic block reduction -> one partial per block
  __shared__ float sdata[BLOCK_THREADS];
  sdata[threadIdx.x] = local;
  __syncthreads();
#pragma unroll
  for (int off = BLOCK_THREADS / 2; off > 0; off >>= 1) {
    if ((int)threadIdx.x < off) sdata[threadIdx.x] += sdata[threadIdx.x + off];
    __syncthreads();
  }
  if (threadIdx.x == 0) partial[blockIdx.x] = sdata[0];
}

__global__ void __launch_bounds__(BLOCK_THREADS)
sgns_reduce_kernel(const float* __restrict__ partial, int n,
                   float* __restrict__ out) {
  __shared__ float sdata[BLOCK_THREADS];
  float s = 0.0f;
  for (int i = threadIdx.x; i < n; i += BLOCK_THREADS) s += partial[i];
  sdata[threadIdx.x] = s;
  __syncthreads();
#pragma unroll
  for (int off = BLOCK_THREADS / 2; off > 0; off >>= 1) {
    if ((int)threadIdx.x < off) sdata[threadIdx.x] += sdata[threadIdx.x + off];
    __syncthreads();
  }
  if (threadIdx.x == 0) out[0] = -sdata[0];
}

extern "C" void kernel_launch(void* const* d_in, const int* in_sizes, int n_in,
                              void* d_out, int out_size, void* d_ws, size_t ws_size,
                              hipStream_t stream) {
  const int*   center    = (const int*)d_in[0];
  const int*   pos       = (const int*)d_in[1];
  const int*   neg       = (const int*)d_in[2];
  const float* in_embed  = (const float*)d_in[3];
  const float* out_embed = (const float*)d_in[4];
  float*       out       = (float*)d_out;

  const int Bn = in_sizes[0];
  const int K  = in_sizes[2] / Bn;            // 10
  const int ngroups = Bn / 16;                // B divisible by 16 (65536)
  const int nblocks = (ngroups + WAVES_PER_BLOCK - 1) / WAVES_PER_BLOCK;

  const float invB  = 1.0f / (float)Bn;
  const float invBK = 1.0f / ((float)Bn * (float)K);

  float* partial = (float*)d_ws;              // nblocks floats of scratch

  sgns_wmma_kernel<<<nblocks, BLOCK_THREADS, 0, stream>>>(
      center, pos, neg, in_embed, out_embed, partial, K, ngroups, invB, invBK);
  sgns_reduce_kernel<<<1, BLOCK_THREADS, 0, stream>>>(partial, nblocks, out);
}
